// ChiStream_25580825215106
// MI455X (gfx1250) — compile-verified, matrix operand order checked by reading, not run on hardware
//
#include <hip/hip_runtime.h>
#include <hip/hip_bf16.h>
#include <stdint.h>

typedef __attribute__((ext_vector_type(16))) _Float16 v16h;
typedef __attribute__((ext_vector_type(8)))  float    v8f;

#define B_     4
#define NTOK   1568
#define C_     512
#define MTOT   (B_*NTOK)     // 6272 = 49*128
#define HEADS_ 8
#define HD     64
#define EPSF   1e-5f

union Frag { v16h h; uint4 q[2]; };

__device__ __forceinline__ v8f zero8() {
  v8f z = {0.f,0.f,0.f,0.f,0.f,0.f,0.f,0.f};
  return z;
}

// ---- gfx1250 async DMA: global -> LDS, 16B per lane, tracked by ASYNCcnt ----
__device__ __forceinline__ void async_b128(void* lds, const void* g) {
  uint32_t lo = (uint32_t)(uintptr_t)lds;          // low 32 bits of generic ptr = LDS offset
  asm volatile("global_load_async_to_lds_b128 %0, %1, off"
               :: "v"(lo), "v"((uint64_t)(uintptr_t)g)
               : "memory");
}
__device__ __forceinline__ void wait_async0() {
  asm volatile("s_wait_asynccnt 0" ::: "memory");
}

// ---------------- tokenize: x[B,C,THW] f32 -> t[B,THW,C] f32 + f16 ----------------
__global__ void __launch_bounds__(256) k_tokenize(const float* __restrict__ x,
                                                  float* __restrict__ tf,
                                                  _Float16* __restrict__ th) {
  int idx = blockIdx.x * 256 + threadIdx.x;      // exactly MTOT*C_ threads
  int c    = idx & (C_ - 1);
  int rest = idx >> 9;
  int pos  = rest % NTOK;
  int b    = rest / NTOK;
  float v = x[((size_t)(b * C_ + c)) * NTOK + pos];
  tf[idx] = v;
  th[idx] = (_Float16)v;
}

// ---------------- f32 -> f16 elementwise ----------------
__global__ void __launch_bounds__(256) k_cast16(const float* __restrict__ s,
                                                _Float16* __restrict__ d, int n) {
  int i = blockIdx.x * 256 + threadIdx.x;
  if (i < n) d[i] = (_Float16)s[i];
}

// ---------------- LayerNorm: one wave per token (C=512) ----------------
__global__ void __launch_bounds__(256) k_ln(const float* __restrict__ tf,
                                            const float* __restrict__ g,
                                            const float* __restrict__ bb,
                                            _Float16* __restrict__ outh) {
  int wv = threadIdx.x >> 5, lane = threadIdx.x & 31;
  int tok = blockIdx.x * 8 + wv;                 // grid 784 -> 6272 tokens exact
  const float* xp = tf + (size_t)tok * C_;
  float vals[16];
  float s = 0.f, s2 = 0.f;
#pragma unroll
  for (int i = 0; i < 16; ++i) {
    float v = xp[lane + i * 32];
    vals[i] = v; s += v; s2 += v * v;
  }
#pragma unroll
  for (int off = 16; off; off >>= 1) {
    s  += __shfl_xor(s,  off, 32);
    s2 += __shfl_xor(s2, off, 32);
  }
  float mean = s * (1.f / C_);
  float var  = s2 * (1.f / C_) - mean * mean;
  float inv  = rsqrtf(var + EPSF);
  _Float16* op = outh + (size_t)tok * C_;
#pragma unroll
  for (int i = 0; i < 16; ++i) {
    int c = lane + i * 32;
    op[c] = (_Float16)((vals[i] - mean) * inv * g[c] + bb[c]);
  }
}

// ---------------- WMMA GEMM: out[M,512] = A[M,512](f16) * W[512,512]^T + bias ----
// Block tile 128x128, 8 waves, wave tile 32x64 (2x4 wmma), async double-buffered LDS.
// MODE 0: store f16 [M,512]
// MODE 1: store f16 transposed per head: vT[b, h, d, pos]
// MODE 2: fp32, relu + bn scale, store NCTHW: out[b, col, pos]
template <int MODE>
__global__ void __launch_bounds__(256) k_gemm(const _Float16* __restrict__ A,
                                              const _Float16* __restrict__ W,
                                              const float* __restrict__ bias,
                                              void* __restrict__ outv,
                                              const float* __restrict__ bng,
                                              const float* __restrict__ bnb) {
  __shared__ _Float16 As[2][128][40];   // 32 halves + 8 pad -> 80B rows, conflict-free
  __shared__ _Float16 Bs[2][128][40];
  int tid  = threadIdx.x;
  int lane = tid & 31, hi = lane >> 4, lm = lane & 15;
  int wv = tid >> 5, wm = wv & 3, wn = wv >> 2;
  int mBase = blockIdx.x * 128;
  int nBase = blockIdx.y * 128;

  v8f acc[2][4];
#pragma unroll
  for (int mi = 0; mi < 2; ++mi)
#pragma unroll
    for (int ni = 0; ni < 4; ++ni) acc[mi][ni] = zero8();

  int lrow = tid >> 2;
  int lcol = (tid & 3) * 8;
  const _Float16* ga = A + ((size_t)(mBase + lrow)) * C_ + lcol;
  const _Float16* gb = W + ((size_t)(nBase + lrow)) * C_ + lcol;

  // prologue: DMA tile 0
  async_b128(&As[0][lrow][lcol],      ga);
  async_b128(&As[0][lrow + 64][lcol], ga + (size_t)64 * C_);
  async_b128(&Bs[0][lrow][lcol],      gb);
  async_b128(&Bs[0][lrow + 64][lcol], gb + (size_t)64 * C_);

  for (int it = 0; it < 16; ++it) {
    int cur = it & 1;
    wait_async0();
    __syncthreads();
    if (it < 15) {
      int nxt = cur ^ 1;
      int k0 = (it + 1) * 32;
      async_b128(&As[nxt][lrow][lcol],      ga + k0);
      async_b128(&As[nxt][lrow + 64][lcol], ga + (size_t)64 * C_ + k0);
      async_b128(&Bs[nxt][lrow][lcol],      gb + k0);
      async_b128(&Bs[nxt][lrow + 64][lcol], gb + (size_t)64 * C_ + k0);
    }
    Frag fa[2], fb[4];
#pragma unroll
    for (int mi = 0; mi < 2; ++mi) {
      int r = wm * 32 + mi * 16 + lm;
      fa[mi].q[0] = *(const uint4*)&As[cur][r][hi * 8];
      fa[mi].q[1] = *(const uint4*)&As[cur][r][hi * 8 + 16];
    }
#pragma unroll
    for (int ni = 0; ni < 4; ++ni) {
      int r = wn * 64 + ni * 16 + lm;
      fb[ni].q[0] = *(const uint4*)&Bs[cur][r][hi * 8];
      fb[ni].q[1] = *(const uint4*)&Bs[cur][r][hi * 8 + 16];
    }
#pragma unroll
    for (int mi = 0; mi < 2; ++mi)
#pragma unroll
      for (int ni = 0; ni < 4; ++ni)
        acc[mi][ni] = __builtin_amdgcn_wmma_f32_16x16x32_f16(
            false, fa[mi].h, false, fb[ni].h, (short)0, acc[mi][ni], false, false);
  }

#pragma unroll
  for (int mi = 0; mi < 2; ++mi)
#pragma unroll
    for (int ni = 0; ni < 4; ++ni) {
      int colg = nBase + wn * 64 + ni * 16 + lm;
      float bcol = bias[colg];
#pragma unroll
      for (int r = 0; r < 8; ++r) {
        int rowg = mBase + wm * 32 + mi * 16 + r + hi * 8;
        float v = acc[mi][ni][r] + bcol;
        if (MODE == 0) {
          ((_Float16*)outv)[(size_t)rowg * C_ + colg] = (_Float16)v;
        } else if (MODE == 1) {
          int bb_ = rowg / NTOK, pos = rowg % NTOK;
          int hh = colg >> 6, dd = colg & 63;
          ((_Float16*)outv)[((size_t)((bb_ * HEADS_ + hh) * HD + dd)) * NTOK + pos] = (_Float16)v;
        } else {
          float z = fmaxf(v, 0.f) * rsqrtf(1.f + EPSF);
          z = z * bng[colg] + bnb[colg];
          int bb_ = rowg / NTOK, pos = rowg % NTOK;
          ((float*)outv)[((size_t)(bb_ * C_ + colg)) * NTOK + pos] = z;
        }
      }
    }
}

// ---------------- flash attention: one (b,h); 8 waves x 16 query rows ----------------
// 64-key chunks (24 full + masked tail), async double-buffered K/V^T staging.
__global__ void __launch_bounds__(256) k_attn(const _Float16* __restrict__ qh,
                                              const _Float16* __restrict__ kh,
                                              const _Float16* __restrict__ vT,
                                              float* __restrict__ tf) {
  __shared__ _Float16 Ks[2][64][72];   // keys x dims
  __shared__ _Float16 Vs[2][64][72];   // dims x keys (V^T rows)
  __shared__ _Float16 Ps[8][16][72];   // per-wave P re-swizzle buffer

  int tid  = threadIdx.x;
  int lane = tid & 31, hi = lane >> 4, lm = lane & 15;
  int wv = tid >> 5;
  int h = blockIdx.y, b = blockIdx.z;
  int qtile = blockIdx.x * 128 + wv * 16;
  int qr = qtile + lm; if (qr > NTOK - 1) qr = NTOK - 1;

  const _Float16* qp = qh + ((size_t)(b * NTOK + qr)) * C_ + h * HD;
  Frag qf[2];
  qf[0].q[0] = *(const uint4*)(qp + hi * 8);
  qf[0].q[1] = *(const uint4*)(qp + hi * 8 + 16);
  qf[1].q[0] = *(const uint4*)(qp + 32 + hi * 8);
  qf[1].q[1] = *(const uint4*)(qp + 32 + hi * 8 + 16);

  float mrun[8], lrun[8];
  v8f O[4];
#pragma unroll
  for (int r = 0; r < 8; ++r) { mrun[r] = -1e30f; lrun[r] = 0.f; }
  O[0] = zero8(); O[1] = zero8(); O[2] = zero8(); O[3] = zero8();
  const float SCALE2 = 0.125f * 1.44269504089f;   // 64^-0.5 * log2(e); use exp2

  int kr  = tid >> 3;              // 0..31
  int kc8 = (tid & 7) * 8;         // 0..56 halves
  const _Float16* kbase = kh + (size_t)b * NTOK * C_ + h * HD;
  const _Float16* vbase = vT + ((size_t)(b * HEADS_ + h)) * HD * NTOK;

  const int NCHUNK = 25;           // 24*64 + masked tail

  // prologue: DMA chunk 0
  {
    int r0 = kr, r1 = kr + 32;     // rows valid for chunk 0
    async_b128(&Ks[0][kr][kc8],      kbase + (size_t)(r0) * C_ + kc8);
    async_b128(&Ks[0][kr + 32][kc8], kbase + (size_t)(r1) * C_ + kc8);
    async_b128(&Vs[0][kr][kc8],      vbase + (size_t)(kr) * NTOK + kc8);
    async_b128(&Vs[0][kr + 32][kc8], vbase + (size_t)(kr + 32) * NTOK + kc8);
  }

  for (int c = 0; c < NCHUNK; ++c) {
    int cur = c & 1;
    int n0 = c * 64;
    wait_async0();
    __syncthreads();
    if (c + 1 < NCHUNK) {
      int nxt = cur ^ 1;
      int nn = (c + 1) * 64;
      int r0 = nn + kr;       if (r0 > NTOK - 1) r0 = NTOK - 1;   // clamp: keep finite
      int r1 = nn + kr + 32;  if (r1 > NTOK - 1) r1 = NTOK - 1;
      int cc = nn + kc8;      if (cc > NTOK - 8) cc = NTOK - 8;
      async_b128(&Ks[nxt][kr][kc8],      kbase + (size_t)r0 * C_ + kc8);
      async_b128(&Ks[nxt][kr + 32][kc8], kbase + (size_t)r1 * C_ + kc8);
      async_b128(&Vs[nxt][kr][kc8],      vbase + (size_t)(kr) * NTOK + cc);
      async_b128(&Vs[nxt][kr + 32][kc8], vbase + (size_t)(kr + 32) * NTOK + cc);
    }

    // S = Q * K^T  (16 x 64): 4 col tiles x 2 contraction steps
    v8f s[4];
    s[0] = zero8(); s[1] = zero8(); s[2] = zero8(); s[3] = zero8();
#pragma unroll
    for (int kc = 0; kc < 2; ++kc) {
#pragma unroll
      for (int j = 0; j < 4; ++j) {
        Frag f;
        f.q[0] = *(const uint4*)&Ks[cur][j * 16 + lm][kc * 32 + hi * 8];
        f.q[1] = *(const uint4*)&Ks[cur][j * 16 + lm][kc * 32 + hi * 8 + 16];
        s[j] = __builtin_amdgcn_wmma_f32_16x16x32_f16(
            false, qf[kc].h, false, f.h, (short)0, s[j], false, false);
      }
    }

    // column-validity mask (only tail chunk has invalid keys)
    bool vm[4];
#pragma unroll
    for (int j = 0; j < 4; ++j) vm[j] = (n0 + j * 16 + lm) < NTOK;

    // online softmax in log2 domain (reg r -> row r + 8*hi; col = j*16 + lm)
#pragma unroll
    for (int r = 0; r < 8; ++r) {
      float a0 = vm[0] ? s[0][r] * SCALE2 : -1e30f;
      float a1 = vm[1] ? s[1][r] * SCALE2 : -1e30f;
      float a2 = vm[2] ? s[2][r] * SCALE2 : -1e30f;
      float a3 = vm[3] ? s[3][r] * SCALE2 : -1e30f;
      float mx = fmaxf(fmaxf(a0, a1), fmaxf(a2, a3));
#pragma unroll
      for (int off = 8; off; off >>= 1) mx = fmaxf(mx, __shfl_xor(mx, off, 32));
      float mnew = fmaxf(mrun[r], mx);
      float corr = exp2f(mrun[r] - mnew);
      float p0 = exp2f(a0 - mnew);
      float p1 = exp2f(a1 - mnew);
      float p2 = exp2f(a2 - mnew);
      float p3 = exp2f(a3 - mnew);
      float ps = (p0 + p1) + (p2 + p3);
#pragma unroll
      for (int off = 8; off; off >>= 1) ps += __shfl_xor(ps, off, 32);
      lrun[r] = lrun[r] * corr + ps;
      mrun[r] = mnew;
#pragma unroll
      for (int t = 0; t < 4; ++t) O[t][r] *= corr;
      Ps[wv][r + hi * 8][lm]      = (_Float16)p0;
      Ps[wv][r + hi * 8][16 + lm] = (_Float16)p1;
      Ps[wv][r + hi * 8][32 + lm] = (_Float16)p2;
      Ps[wv][r + hi * 8][48 + lm] = (_Float16)p3;
    }
    // intra-wave LDS RAW across lanes: wait ds stores, reload as A-fragments
    asm volatile("s_wait_dscnt 0" ::: "memory");

#pragma unroll
    for (int kc = 0; kc < 2; ++kc) {
      Frag pf;
      pf.q[0] = *(const uint4*)&Ps[wv][lm][kc * 32 + hi * 8];
      pf.q[1] = *(const uint4*)&Ps[wv][lm][kc * 32 + hi * 8 + 16];
#pragma unroll
      for (int t = 0; t < 4; ++t) {
        Frag vf;
        vf.q[0] = *(const uint4*)&Vs[cur][t * 16 + lm][kc * 32 + hi * 8];
        vf.q[1] = *(const uint4*)&Vs[cur][t * 16 + lm][kc * 32 + hi * 8 + 16];
        O[t] = __builtin_amdgcn_wmma_f32_16x16x32_f16(
            false, pf.h, false, vf.h, (short)0, O[t], false, false);
      }
    }
  }

#pragma unroll
  for (int r = 0; r < 8; ++r) {
    float inv = lrun[r] > 0.f ? 1.f / lrun[r] : 0.f;
    int rowg = qtile + r + hi * 8;
    if (rowg < NTOK) {
#pragma unroll
      for (int t = 0; t < 4; ++t) {
        int cg = h * HD + t * 16 + lm;
        size_t o = ((size_t)(b * NTOK + rowg)) * C_ + cg;
        tf[o] += O[t][r] * inv;    // residual add; (row,col) owned exclusively
      }
    }
  }
}

// =====================================================================
extern "C" void kernel_launch(void* const* d_in, const int* in_sizes, int n_in,
                              void* d_out, int out_size, void* d_ws, size_t ws_size,
                              hipStream_t stream) {
  (void)in_sizes; (void)n_in; (void)out_size; (void)ws_size;
  const float* x1   = (const float*)d_in[0];
  const float* x2   = (const float*)d_in[1];
  const float* Wq1  = (const float*)d_in[2];
  const float* bq1  = (const float*)d_in[3];
  const float* Wk1  = (const float*)d_in[4];
  const float* bk1  = (const float*)d_in[5];
  const float* Wv1  = (const float*)d_in[6];
  const float* bv1  = (const float*)d_in[7];
  const float* ln1g = (const float*)d_in[8];
  const float* ln1b = (const float*)d_in[9];
  const float* Wq2  = (const float*)d_in[10];
  const float* bq2  = (const float*)d_in[11];
  const float* Wk2  = (const float*)d_in[12];
  const float* bk2  = (const float*)d_in[13];
  const float* Wv2  = (const float*)d_in[14];
  const float* bv2  = (const float*)d_in[15];
  const float* ln2g = (const float*)d_in[16];
  const float* ln2b = (const float*)d_in[17];
  const float* o1w  = (const float*)d_in[18];
  const float* o1b  = (const float*)d_in[19];
  const float* bn1g = (const float*)d_in[20];
  const float* bn1b = (const float*)d_in[21];
  const float* o2w  = (const float*)d_in[22];
  const float* o2b  = (const float*)d_in[23];
  const float* bn2g = (const float*)d_in[24];
  const float* bn2b = (const float*)d_in[25];

  char* ws = (char*)d_ws;
  size_t off = 0;
  auto alloc = [&](size_t bytes) -> void* {
    void* p = ws + off;
    off += (bytes + 255) & ~(size_t)255;
    return p;
  };
  const size_t F32TOK = (size_t)MTOT * C_ * 4;
  const size_t F16TOK = (size_t)MTOT * C_ * 2;
  float*    t1f  = (float*)alloc(F32TOK);
  float*    t2f  = (float*)alloc(F32TOK);
  _Float16* t1h  = (_Float16*)alloc(F16TOK);
  _Float16* t2h  = (_Float16*)alloc(F16TOK);
  _Float16* n1h  = (_Float16*)alloc(F16TOK);
  _Float16* n2h  = (_Float16*)alloc(F16TOK);
  _Float16* q1h  = (_Float16*)alloc(F16TOK);
  _Float16* k1h  = (_Float16*)alloc(F16TOK);
  _Float16* q2h  = (_Float16*)alloc(F16TOK);
  _Float16* k2h  = (_Float16*)alloc(F16TOK);
  _Float16* v1T  = (_Float16*)alloc(F16TOK);   // [B, H, 64, NTOK]
  _Float16* v2T  = (_Float16*)alloc(F16TOK);
  const size_t W3 = (size_t)3 * C_ * C_;
  _Float16* wq1h = (_Float16*)alloc(W3 * 2);
  _Float16* wk1h = (_Float16*)alloc(W3 * 2);
  _Float16* wv1h = (_Float16*)alloc(W3 * 2);
  _Float16* wq2h = (_Float16*)alloc(W3 * 2);
  _Float16* wk2h = (_Float16*)alloc(W3 * 2);
  _Float16* wv2h = (_Float16*)alloc(W3 * 2);
  _Float16* wo1h = (_Float16*)alloc((size_t)C_ * C_ * 2);
  _Float16* wo2h = (_Float16*)alloc((size_t)C_ * C_ * 2);

  dim3 blk(256);
  const int NELEM = MTOT * C_;                 // 3,211,264 = 12544*256
  k_tokenize<<<NELEM / 256, blk, 0, stream>>>(x1, t1f, t1h);
  k_tokenize<<<NELEM / 256, blk, 0, stream>>>(x2, t2f, t2h);

  const int WN3 = (int)W3;
  k_cast16<<<(WN3 + 255) / 256, blk, 0, stream>>>(Wq1, wq1h, WN3);
  k_cast16<<<(WN3 + 255) / 256, blk, 0, stream>>>(Wk1, wk1h, WN3);
  k_cast16<<<(WN3 + 255) / 256, blk, 0, stream>>>(Wv1, wv1h, WN3);
  k_cast16<<<(WN3 + 255) / 256, blk, 0, stream>>>(Wq2, wq2h, WN3);
  k_cast16<<<(WN3 + 255) / 256, blk, 0, stream>>>(Wk2, wk2h, WN3);
  k_cast16<<<(WN3 + 255) / 256, blk, 0, stream>>>(Wv2, wv2h, WN3);
  const int WN1 = C_ * C_;
  k_cast16<<<(WN1 + 255) / 256, blk, 0, stream>>>(o1w, wo1h, WN1);
  k_cast16<<<(WN1 + 255) / 256, blk, 0, stream>>>(o2w, wo2h, WN1);

  dim3 ggrid(49, 4);            // 6272/128 x 512/128
  dim3 agrid(13, HEADS_, B_);   // ceil(1568/128)

  for (int s = 0; s < 3; ++s) {
    size_t wo = (size_t)s * C_ * C_;
    int bo = s * C_;
    k_ln<<<784, blk, 0, stream>>>(t1f, ln1g + bo, ln1b + bo, n1h);
    k_ln<<<784, blk, 0, stream>>>(t2f, ln2g + bo, ln2b + bo, n2h);

    k_gemm<0><<<ggrid, blk, 0, stream>>>(n1h, wq1h + wo, bq1 + bo, q1h, nullptr, nullptr);
    k_gemm<0><<<ggrid, blk, 0, stream>>>(t2h, wk1h + wo, bk1 + bo, k1h, nullptr, nullptr);
    k_gemm<1><<<ggrid, blk, 0, stream>>>(t2h, wv1h + wo, bv1 + bo, v1T, nullptr, nullptr);
    k_gemm<0><<<ggrid, blk, 0, stream>>>(n2h, wq2h + wo, bq2 + bo, q2h, nullptr, nullptr);
    k_gemm<0><<<ggrid, blk, 0, stream>>>(t1h, wk2h + wo, bk2 + bo, k2h, nullptr, nullptr);
    k_gemm<1><<<ggrid, blk, 0, stream>>>(t1h, wv2h + wo, bv2 + bo, v2T, nullptr, nullptr);

    k_attn<<<agrid, blk, 0, stream>>>(q1h, k1h, v1T, t1f);
    k_attn<<<agrid, blk, 0, stream>>>(q2h, k2h, v2T, t2f);

    k_cast16<<<NELEM / 256, blk, 0, stream>>>(t1f, t1h, NELEM);
    k_cast16<<<NELEM / 256, blk, 0, stream>>>(t2f, t2h, NELEM);
  }

  float* z1 = (float*)d_out;
  float* z2 = z1 + (size_t)B_ * C_ * NTOK;
  k_gemm<2><<<ggrid, blk, 0, stream>>>(t1h, wo1h, o1b, z1, bn1g, bn1b);
  k_gemm<2><<<ggrid, blk, 0, stream>>>(t2h, wo2h, o2b, z2, bn2g, bn2b);
}
